// PointNet_SA_Layer_2181843386552
// MI455X (gfx1250) — matrix operand order repaired and checked by reading, not compiled, not run on hardware
//
#include <hip/hip_runtime.h>
#include <hip/hip_bf16.h>

#define B_ 16
#define N_ 16384
#define S_ 2048
#define K_ 32
#define D_ 64

typedef __attribute__((ext_vector_type(16))) _Float16 v16h;
typedef __attribute__((ext_vector_type(8)))  _Float16 v8h;
typedef __attribute__((ext_vector_type(8)))  float    v8f;

union F16x16 { v16h v; v8h h[2]; };

// A-fragment for 16x32 f16 tile from row-major [rows][strid] f16 matrix.
// lane<16: M=lane, K = {0..7,16..23}; lane>=16: M=lane-16, K = {8..15,24..31}
__device__ __forceinline__ v16h lda(const _Float16* base, int strid, int row, int kt, int hi) {
  F16x16 f;
  const _Float16* r = base + row * strid + kt * 32;
  f.h[0] = *(const v8h*)(r + hi * 8);
  f.h[1] = *(const v8h*)(r + 16 + hi * 8);
  return f.v;
}
// B-fragment (32x16): lane<16: N=lane, K = {0..7,16..23}; lane>=16: N=lane-16, K = {8..15,24..31}.
// Weight matrix stored row-major [outCh][strid] so B column n == weight row n.
__device__ __forceinline__ v16h ldb(const _Float16* base, int strid, int n, int kt, int hi) {
  F16x16 f;
  const _Float16* r = base + n * strid + kt * 32;
  f.h[0] = *(const v8h*)(r + hi * 8);
  f.h[1] = *(const v8h*)(r + 16 + hi * 8);
  return f.v;
}

// ---------------- Kernel 1: farthest point sampling (1 block per batch) ----
#define PPT 16  // 16384 points / 1024 threads
__global__ __launch_bounds__(1024) void fps_kernel(const float* __restrict__ xyz,
                                                   float* __restrict__ sxyz) {
  const int b = blockIdx.x;
  const int tid = threadIdx.x;
  const float* xb = xyz + (size_t)b * 3 * N_;
  float px[PPT], py[PPT], pz[PPT], mind[PPT];
#pragma unroll
  for (int j = 0; j < PPT; ++j) {
    int i = j * 1024 + tid;                 // stride-1 across threads: coalesced
    px[j] = xb[i];
    py[j] = xb[N_ + i];
    pz[j] = xb[2 * N_ + i];
    mind[j] = 1e10f;
  }
  __shared__ float bx, by, bz;
  __shared__ float rv[32];
  __shared__ int   ri[32];
  __shared__ int   bidx;
  if (tid == 0) { bx = px[0]; by = py[0]; bz = pz[0]; }  // start index 0
  __syncthreads();

  int last = 0;
  for (int s = 0; s < S_; ++s) {
    const float lx = bx, ly = by, lz = bz;
    if (tid == 0) {   // record current "last" (reference emits old last each step)
      sxyz[((size_t)b * S_ + s) * 3 + 0] = lx;
      sxyz[((size_t)b * S_ + s) * 3 + 1] = ly;
      sxyz[((size_t)b * S_ + s) * 3 + 2] = lz;
    }
    float bestv = -1.0f; int besti = 0;
#pragma unroll
    for (int j = 0; j < PPT; ++j) {
      int i = j * 1024 + tid;
      float dx = px[j] - lx, dy = py[j] - ly, dz = pz[j] - lz;
      float d = dx * dx + dy * dy + dz * dz;
      float m = fminf(mind[j], d);
      mind[j] = m;
      if (m > bestv) { bestv = m; besti = i; }   // j ascending -> lowest idx on tie
    }
#pragma unroll
    for (int off = 16; off > 0; off >>= 1) {
      float ov = __shfl_xor(bestv, off, 32);
      int   oi = __shfl_xor(besti, off, 32);
      if (ov > bestv || (ov == bestv && oi < besti)) { bestv = ov; besti = oi; }
    }
    if ((tid & 31) == 0) { rv[tid >> 5] = bestv; ri[tid >> 5] = besti; }
    __syncthreads();
    if (tid < 32) {
      float v = rv[tid]; int i2 = ri[tid];
#pragma unroll
      for (int off = 16; off > 0; off >>= 1) {
        float ov = __shfl_xor(v, off, 32);
        int   oi = __shfl_xor(i2, off, 32);
        if (ov > v || (ov == v && oi < i2)) { v = ov; i2 = oi; }
      }
      if (tid == 0) bidx = i2;
    }
    __syncthreads();
    last = bidx;
    if (tid == (last & 1023)) {             // owner publishes coords of new pick
      int jj = last >> 10;
#pragma unroll
      for (int j = 0; j < PPT; ++j)
        if (j == jj) { bx = px[j]; by = py[j]; bz = pz[j]; }
    }
    __syncthreads();
  }
}

// ---------------- Kernel 2: points[b, :, n<2048] -> f16 [b, n, 64] ---------
__global__ __launch_bounds__(256) void transpose_kernel(const float* __restrict__ points,
                                                        _Float16* __restrict__ ptsT) {
  const int t = blockIdx.x * 256 + threadIdx.x;  // 0 .. B*2048-1
  const int b = t >> 11, n = t & 2047;
  const float* pb = points + (size_t)b * D_ * N_ + n;
  v8h* dst = (v8h*)(ptsT + (size_t)t * 64);
#pragma unroll
  for (int q = 0; q < 8; ++q) {
    v8h v;
#pragma unroll
    for (int e = 0; e < 8; ++e) v[e] = (_Float16)pb[(size_t)(q * 8 + e) * N_];
    dst[q] = v;
  }
}

// ---------------- Kernel 3: knn over sampled points (1 wave per row) -------
__global__ __launch_bounds__(256) void knn_kernel(const float* __restrict__ sxyz,
                                                  int* __restrict__ nn) {
  __shared__ float sx[S_], sy[S_], sz[S_], sq[S_];
  const int b = blockIdx.y;
  const int tid = threadIdx.x;
  for (int i = tid; i < S_; i += 256) {
    float x = sxyz[((size_t)b * S_ + i) * 3 + 0];
    float y = sxyz[((size_t)b * S_ + i) * 3 + 1];
    float z = sxyz[((size_t)b * S_ + i) * 3 + 2];
    sx[i] = x; sy[i] = y; sz[i] = z; sq[i] = x * x + y * y + z * z;
  }
  __syncthreads();
  const int wave = tid >> 5, lane = tid & 31;
  const int s = blockIdx.x * 8 + wave;
  const float cx = sx[s], cy = sy[s], cz = sz[s], csq = sq[s];
  float d[64];
#pragma unroll
  for (int j = 0; j < 64; ++j) {
    int t = j * 32 + lane;
    d[j] = csq + sq[t] - 2.0f * (cx * sx[t] + cy * sy[t] + cz * sz[t]);
  }
  int myidx = 0;
#pragma unroll 1
  for (int r = 0; r < K_; ++r) {
    float bv = d[0]; int bj = 0;
#pragma unroll
    for (int j = 1; j < 64; ++j)
      if (d[j] < bv) { bv = d[j]; bj = j; }     // j ascending -> lowest t on tie
    int bt = bj * 32 + lane;
#pragma unroll
    for (int off = 16; off > 0; off >>= 1) {
      float ov = __shfl_xor(bv, off, 32);
      int   ot = __shfl_xor(bt, off, 32);
      if (ov < bv || (ov == bv && ot < bt)) { bv = ov; bt = ot; }
    }
    const int wj = bt >> 5;
    if ((bt & 31) == lane) {
#pragma unroll
      for (int j = 0; j < 64; ++j)
        if (j == wj) d[j] = 3e38f;
    }
    if (r == lane) myidx = bt;                  // lane r keeps r-th nearest
  }
  nn[((size_t)b * S_ + s) * K_ + lane] = myidx; // coalesced 128B store
}

// ---------------- Kernel 4: gather + WMMA MLP (1 wave per centroid) --------
__global__ __launch_bounds__(128) void mlp_kernel(
    const _Float16* __restrict__ ptsT, const float* __restrict__ sxyz,
    const int* __restrict__ nn,
    const float* __restrict__ W0, const float* __restrict__ b0,
    const float* __restrict__ W1, const float* __restrict__ b1,
    const float* __restrict__ W2, const float* __restrict__ b2,
    float* __restrict__ out) {
  __shared__ __align__(16) _Float16 w0s[64][96];   // 67 in-ch padded to 96
  __shared__ __align__(16) _Float16 w1s[64][64];
  __shared__ __align__(16) _Float16 w2s[128][64];
  __shared__ float b0s[64], b1s[64], b2s[128];
  __shared__ __align__(16) _Float16 gs[4][32][96];
  __shared__ __align__(16) _Float16 h0s[4][32][64];
  __shared__ __align__(16) _Float16 h1s[4][32][64];
  __shared__ __align__(16) float    sts[4][32][16];

  const int tid = threadIdx.x;
  for (int i = tid; i < 64 * 96; i += 128) {
    int o = i / 96, k = i - o * 96;
    w0s[o][k] = (k < 67) ? (_Float16)W0[o * 67 + k] : (_Float16)0.0f;
  }
  for (int i = tid; i < 64 * 64; i += 128) (&w1s[0][0])[i] = (_Float16)W1[i];
  for (int i = tid; i < 128 * 64; i += 128) (&w2s[0][0])[i] = (_Float16)W2[i];
  if (tid < 64) { b0s[tid] = b0[tid]; b1s[tid] = b1[tid]; }
  if (tid < 128) b2s[tid] = b2[tid];
  __syncthreads();

  const int wave = tid >> 5, lane = tid & 31;
  const int l15 = lane & 15, hi = lane >> 4;
  const int gid = blockIdx.x * 4 + wave;
  const int b = gid >> 11, s = gid & (S_ - 1);

  _Float16 (*gw)[96]  = gs[wave];
  _Float16 (*h0w)[64] = h0s[wave];
  _Float16 (*h1w)[64] = h1s[wave];
  float    (*stw)[16] = sts[wave];

  // ---- gather: lane k builds input row k (64 feat + 3 rel-xyz + pad) ----
  const int nidx = nn[((size_t)(b * S_ + s)) * K_ + lane];
  const _Float16* srcf = ptsT + ((size_t)(b * S_ + nidx)) * 64;  // contiguous 128B
#pragma unroll
  for (int q = 0; q < 8; ++q)
    *(v8h*)&gw[lane][q * 8] = *(const v8h*)&srcf[q * 8];
  const float cx = sxyz[((size_t)(b * S_ + s)) * 3 + 0];
  const float cy = sxyz[((size_t)(b * S_ + s)) * 3 + 1];
  const float cz = sxyz[((size_t)(b * S_ + s)) * 3 + 2];
  const float nx = sxyz[((size_t)(b * S_ + nidx)) * 3 + 0];
  const float ny = sxyz[((size_t)(b * S_ + nidx)) * 3 + 1];
  const float nz = sxyz[((size_t)(b * S_ + nidx)) * 3 + 2];
  gw[lane][64] = (_Float16)(nx - cx);
  gw[lane][65] = (_Float16)(ny - cy);
  gw[lane][66] = (_Float16)(nz - cz);
#pragma unroll
  for (int k = 67; k < 96; ++k) gw[lane][k] = (_Float16)0.0f;
  __builtin_amdgcn_wave_barrier();   // same-wave LDS is in-order; block compiler reordering

  // ---- layer 0: [32x96] x W0^T[96x64] -> relu -> h0 [32x64] ----
#pragma unroll
  for (int mt = 0; mt < 2; ++mt) {
    v16h a0 = lda(&gw[0][0], 96, mt * 16 + l15, 0, hi);
    v16h a1 = lda(&gw[0][0], 96, mt * 16 + l15, 1, hi);
    v16h a2 = lda(&gw[0][0], 96, mt * 16 + l15, 2, hi);
#pragma unroll
    for (int nt = 0; nt < 4; ++nt) {
      v8f acc = {0.f, 0.f, 0.f, 0.f, 0.f, 0.f, 0.f, 0.f};
      acc = __builtin_amdgcn_wmma_f32_16x16x32_f16(false, a0, false,
              ldb(&w0s[0][0], 96, nt * 16 + l15, 0, hi), (short)0, acc, false, false);
      acc = __builtin_amdgcn_wmma_f32_16x16x32_f16(false, a1, false,
              ldb(&w0s[0][0], 96, nt * 16 + l15, 1, hi), (short)0, acc, false, false);
      acc = __builtin_amdgcn_wmma_f32_16x16x32_f16(false, a2, false,
              ldb(&w0s[0][0], 96, nt * 16 + l15, 2, hi), (short)0, acc, false, false);
      const float bv = b0s[nt * 16 + l15];
#pragma unroll
      for (int v = 0; v < 8; ++v)
        h0w[mt * 16 + v + 8 * hi][nt * 16 + l15] = (_Float16)fmaxf(acc[v] + bv, 0.0f);
    }
  }
  __builtin_amdgcn_wave_barrier();

  // ---- layer 1: [32x64] x W1^T[64x64] -> relu -> h1 [32x64] ----
#pragma unroll
  for (int mt = 0; mt < 2; ++mt) {
    v16h a0 = lda(&h0w[0][0], 64, mt * 16 + l15, 0, hi);
    v16h a1 = lda(&h0w[0][0], 64, mt * 16 + l15, 1, hi);
#pragma unroll
    for (int nt = 0; nt < 4; ++nt) {
      v8f acc = {0.f, 0.f, 0.f, 0.f, 0.f, 0.f, 0.f, 0.f};
      acc = __builtin_amdgcn_wmma_f32_16x16x32_f16(false, a0, false,
              ldb(&w1s[0][0], 64, nt * 16 + l15, 0, hi), (short)0, acc, false, false);
      acc = __builtin_amdgcn_wmma_f32_16x16x32_f16(false, a1, false,
              ldb(&w1s[0][0], 64, nt * 16 + l15, 1, hi), (short)0, acc, false, false);
      const float bv = b1s[nt * 16 + l15];
#pragma unroll
      for (int v = 0; v < 8; ++v)
        h1w[mt * 16 + v + 8 * hi][nt * 16 + l15] = (_Float16)fmaxf(acc[v] + bv, 0.0f);
    }
  }
  __builtin_amdgcn_wave_barrier();

  // ---- layer 2: [32x64] x W2^T[64x128] + bias; stage 16 ch -> coalesced store
  v16h A2[2][2];
#pragma unroll
  for (int mt = 0; mt < 2; ++mt)
#pragma unroll
    for (int kt = 0; kt < 2; ++kt)
      A2[mt][kt] = lda(&h1w[0][0], 64, mt * 16 + l15, kt, hi);
#pragma unroll
  for (int nt = 0; nt < 8; ++nt) {
    const float bv = b2s[nt * 16 + l15];
#pragma unroll
    for (int mt = 0; mt < 2; ++mt) {
      v8f acc = {0.f, 0.f, 0.f, 0.f, 0.f, 0.f, 0.f, 0.f};
      acc = __builtin_amdgcn_wmma_f32_16x16x32_f16(false, A2[mt][0], false,
              ldb(&w2s[0][0], 64, nt * 16 + l15, 0, hi), (short)0, acc, false, false);
      acc = __builtin_amdgcn_wmma_f32_16x16x32_f16(false, A2[mt][1], false,
              ldb(&w2s[0][0], 64, nt * 16 + l15, 1, hi), (short)0, acc, false, false);
#pragma unroll
      for (int v = 0; v < 8; ++v)
        stw[mt * 16 + v + 8 * hi][l15] = acc[v] + bv;
    }
    __builtin_amdgcn_wave_barrier();
    // out[b][o][s][k]: 32 contiguous floats per channel row (128B stores)
    const size_t obase = (((size_t)b * 128 + nt * 16) * S_ + s) * K_;
#pragma unroll
    for (int o = 0; o < 16; ++o)
      out[obase + (size_t)o * (S_ * K_) + lane] = stw[lane][o];
    __builtin_amdgcn_wave_barrier();
  }
}

extern "C" void kernel_launch(void* const* d_in, const int* in_sizes, int n_in,
                              void* d_out, int out_size, void* d_ws, size_t ws_size,
                              hipStream_t stream) {
  (void)in_sizes; (void)n_in; (void)out_size; (void)ws_size;
  const float* xyz    = (const float*)d_in[0];
  const float* points = (const float*)d_in[1];
  const float* W0 = (const float*)d_in[2];
  const float* b0 = (const float*)d_in[3];
  const float* W1 = (const float*)d_in[4];
  const float* b1 = (const float*)d_in[5];
  const float* W2 = (const float*)d_in[6];
  const float* b2 = (const float*)d_in[7];
  float* out = (float*)d_out;

  // workspace layout
  float*    sxyz = (float*)d_ws;                                   // 16*2048*3 f32 = 384KB
  int*      nn   = (int*)((char*)d_ws + (size_t)393216);           // 16*2048*32 i32 = 4MB
  _Float16* ptsT = (_Float16*)((char*)d_ws + (size_t)393216 + (size_t)4194304); // 4MB

  hipLaunchKernelGGL(fps_kernel, dim3(B_), dim3(1024), 0, stream, xyz, sxyz);
  hipLaunchKernelGGL(transpose_kernel, dim3((B_ * S_) / 256), dim3(256), 0, stream, points, ptsT);
  hipLaunchKernelGGL(knn_kernel, dim3(S_ / 8, B_), dim3(256), 0, stream, sxyz, nn);
  hipLaunchKernelGGL(mlp_kernel, dim3((B_ * S_) / 4), dim3(128), 0, stream,
                     ptsT, sxyz, nn, W0, b0, W1, b1, W2, b2, out);
}
